// BDHEncoder_79877801771616
// MI455X (gfx1250) — matrix operand order repaired and checked by reading, not compiled, not run on hardware
//
#include <hip/hip_runtime.h>
#include <hip/hip_bf16.h>
#include <cmath>

// ---------------------------------------------------------------------------
// CDNA5 / gfx1250 BDH encoder.
//  - All matmuls on v_wmma_f32_16x16x32_f16 (wave32, f32 accumulate).
//  - Attention reassociated: (qk qk^T) v == qk (qk^T v)  (no softmax).
//  - f16 end-to-end matmul datapath: weights converted once at launch,
//    activations produced in f16 by their producer kernels. GEMM hot loop
//    is pure 16B f16 copies (global_load_b128 -> ds_store_b128), no cvt.
//  - Software-pipelined (ping-pong LDS) GEMM & attn1: one barrier per K-step,
//    global loads for step k+1 in flight under the WMMAs of step k.
// ---------------------------------------------------------------------------

typedef __attribute__((ext_vector_type(16))) _Float16 v16h;
typedef __attribute__((ext_vector_type(8)))  _Float16 v8h;
typedef __attribute__((ext_vector_type(4)))  _Float16 v4h;
typedef __attribute__((ext_vector_type(8)))  float    v8f;

__device__ __forceinline__ v8f wmma_f16(v16h a, v16h b, v8f c) {
    return __builtin_amdgcn_wmma_f32_16x16x32_f16(false, a, false, b,
                                                  (short)0, c, false, false);
}

__device__ __forceinline__ v4h pack4(float4 a) {
    v4h o;
    o[0] = (_Float16)a.x; o[1] = (_Float16)a.y; o[2] = (_Float16)a.z; o[3] = (_Float16)a.w;
    return o;
}

// A-fragment (16x32 f16) from LDS row, contiguous k order (k = 16h + s), matching frag_b.
__device__ __forceinline__ v16h frag_a(const _Float16* row, int lane) {
    const int kb = (lane < 16) ? 0 : 16;
    v16h a;
    #pragma unroll
    for (int j = 0; j < 16; ++j) a[j] = row[kb + j];
    return a;
}
// B-fragment (32x16 f16): row = K-contiguous data for this lane's column.
__device__ __forceinline__ v16h frag_b(const _Float16* row, int lane) {
    const int kb = (lane < 16) ? 0 : 16;
    v16h b;
    #pragma unroll
    for (int j = 0; j < 16; ++j) b[j] = row[kb + j];
    return b;
}

// ===========================================================================
// f32 -> f16 bulk convert (weights at launch)
// ===========================================================================
__global__ void cvt_f16_kernel(const float* __restrict__ src,
                               _Float16* __restrict__ dst, int n4)
{
    int i = blockIdx.x * blockDim.x + threadIdx.x;
    if (i >= n4) return;
    float4 f = ((const float4*)src)[i];
    ((v4h*)dst)[i] = pack4(f);
}

// ===========================================================================
// WMMA GEMM (f16 x f16 -> f32 acc): Y = act( Xh[M,K] @ Wh^T + bias )
// Wh is [N,K]. act: 0 none, 1 relu, 2 sigmoid.
// Output: Yh (f16) if non-null, else Yf (f32). gate!=null: Yf = resid+gate*v.
// Block = 256 threads = 8 waves; tile 128(M) x 64(N); 2x2 wmma per wave.
// Ping-pong LDS double buffering, one barrier per K-step.
// ===========================================================================
__global__ void wmma_gemm_kernel(const _Float16* __restrict__ Xh,
                                 const _Float16* __restrict__ Wh,
                                 const float* __restrict__ bias,
                                 float* __restrict__ Yf,
                                 _Float16* __restrict__ Yh,
                                 int M, int N, int K, int act,
                                 const float* __restrict__ gate,
                                 const float* __restrict__ resid)
{
    __shared__ _Float16 As[2][128][40];   // 80B row stride (16B aligned)
    __shared__ _Float16 Ws[2][64][40];

    const int tid  = threadIdx.x;
    const int wave = tid >> 5;
    const int lane = tid & 31;
    const int l15  = lane & 15;
    const int m0   = blockIdx.x * 128;
    const int n0   = blockIdx.y * 64;
    const int wm   = (wave >> 1) * 32;   // 0,32,64,96
    const int wn   = (wave & 1) * 32;    // 0,32

    const int ar = tid >> 1;             // A stage: row 0..127, 16 halves each
    const int ac = (tid & 1) * 16;
    const int wr = tid >> 2;             // W stage: row 0..63, 8 halves each
    const int wc = (tid & 3) * 8;
    const bool aok = (m0 + ar < M);
    const bool wok = (n0 + wr < N);
    const _Float16* arow = Xh + (size_t)(m0 + ar) * K + ac;
    const _Float16* wrow = Wh + (size_t)(n0 + wr) * K + wc;

    v8f acc00 = {}, acc01 = {}, acc10 = {}, acc11 = {};
    v8h ra0, ra1, rw0;

    auto load_regs = [&](int kk) {
        v8h a0 = {}, a1 = {}, w0 = {};
        if (aok) {
            const v8h* p = (const v8h*)(arow + kk);
            a0 = p[0]; a1 = p[1];
        }
        if (wok) w0 = *(const v8h*)(wrow + kk);
        ra0 = a0; ra1 = a1; rw0 = w0;
    };
    auto store_lds = [&](int buf) {
        *(v8h*)&As[buf][ar][ac]     = ra0;
        *(v8h*)&As[buf][ar][ac + 8] = ra1;
        *(v8h*)&Ws[buf][wr][wc]     = rw0;
    };

    load_regs(0);
    store_lds(0);
    __syncthreads();

    for (int k0 = 0; k0 < K; k0 += 32) {
        const int cur = (k0 >> 5) & 1;
        const bool more = (k0 + 32 < K);
        if (more) load_regs(k0 + 32);            // loads fly under the WMMAs
        if (k0 + 64 < K && aok)
            __builtin_prefetch((const void*)(arow + k0 + 64), 0, 1);

        const v16h a0 = frag_a(&As[cur][wm + l15][0], lane);
        const v16h a1 = frag_a(&As[cur][wm + 16 + l15][0], lane);
        const v16h b0 = frag_b(&Ws[cur][wn + l15][0], lane);
        const v16h b1 = frag_b(&Ws[cur][wn + 16 + l15][0], lane);
        acc00 = wmma_f16(a0, b0, acc00);
        acc01 = wmma_f16(a0, b1, acc01);
        acc10 = wmma_f16(a1, b0, acc10);
        acc11 = wmma_f16(a1, b1, acc11);

        if (more) store_lds(cur ^ 1);            // waits loads, fills next buf
        __syncthreads();
    }

    // ---- epilogue: C/D layout VGPR r -> M = r (lanes 0-15) / 8+r ----
    #pragma unroll
    for (int sm = 0; sm < 2; ++sm) {
        #pragma unroll
        for (int sn = 0; sn < 2; ++sn) {
            const v8f& acc = sm ? (sn ? acc11 : acc10) : (sn ? acc01 : acc00);
            const int n = n0 + wn + sn * 16 + l15;
            const float bn = bias ? bias[n] : 0.0f;
            #pragma unroll
            for (int r = 0; r < 8; ++r) {
                int m = m0 + wm + sm * 16 + ((lane < 16) ? r : 8 + r);
                if (m < M) {
                    float v = acc[r] + bn;
                    if (act == 1)      v = fmaxf(v, 0.0f);
                    else if (act == 2) v = 1.0f / (1.0f + __expf(-v));
                    size_t idx = (size_t)m * N + n;
                    if (Yh) {
                        Yh[idx] = (_Float16)v;
                    } else {
                        if (gate) v = resid[idx] + gate[idx] * v;
                        Yf[idx] = v;
                    }
                }
            }
        }
    }
}

// ===========================================================================
// Attention stage 1: S[b,h] = scale * QK_h^T @ V_h   (hd x hd, K = Ntok)
// One block per (b,h). f16 inputs staged TRANSPOSED into ping-pong LDS.
// Each wave owns up to 8 16x16 tiles of S; one barrier per K-step.
// ===========================================================================
__global__ void attn_qkT_v_kernel(const _Float16* __restrict__ QK,
                                  const _Float16* __restrict__ V,
                                  float* __restrict__ S,
                                  int B, int Nt, int D, int heads, int hd,
                                  float scale)
{
    __shared__ _Float16 Qt[2][128][40];   // [m (hd)][k (32)] transposed chunk
    __shared__ _Float16 Vt[2][128][40];   // [n (hd)][k (32)]

    const int bh = blockIdx.x;
    const int b  = bh / heads;
    const int h  = bh - b * heads;
    const int tid  = threadIdx.x;
    const int wave = tid >> 5;
    const int lane = tid & 31;
    const int l15  = lane & 15;
    const int tn = hd >> 4;
    const int tiles = tn * tn;

    const _Float16* Qb = QK + (size_t)b * Nt * D + h * hd;
    const _Float16* Vb = V  + (size_t)b * Nt * D + h * hd;

    v8f accs[8] = {};
    const int cpr = hd >> 3;              // 8-half chunks per token row
    const int CH  = 32 * cpr;             // chunks per K-step (<= 512)
    v8h qreg[2], vreg[2];

    auto load_regs = [&](int k0) {
        #pragma unroll
        for (int it = 0; it < 2; ++it) {
            int c = tid + it * 256;
            v8h q = {}, v = {};
            if (c < CH) {
                int r  = c / cpr;
                int cc = (c - r * cpr) * 8;
                int k  = k0 + r;
                if (k < Nt) {
                    q = *(const v8h*)(Qb + (size_t)k * D + cc);
                    v = *(const v8h*)(Vb + (size_t)k * D + cc);
                }
            }
            qreg[it] = q; vreg[it] = v;
        }
    };
    auto store_lds = [&](int buf) {
        #pragma unroll
        for (int it = 0; it < 2; ++it) {
            int c = tid + it * 256;
            if (c < CH) {
                int r  = c / cpr;
                int cc = (c - r * cpr) * 8;
                #pragma unroll
                for (int j = 0; j < 8; ++j) {
                    Qt[buf][cc + j][r] = qreg[it][j];
                    Vt[buf][cc + j][r] = vreg[it][j];
                }
            }
        }
    };

    load_regs(0);
    store_lds(0);
    __syncthreads();

    for (int k0 = 0; k0 < Nt; k0 += 32) {
        const int cur = (k0 >> 5) & 1;
        const bool more = (k0 + 32 < Nt);
        if (more) load_regs(k0 + 32);

        #pragma unroll
        for (int i = 0; i < 8; ++i) {
            int t = wave + i * 8;
            if (t < tiles) {                 // wave-uniform
                int mi = (t / tn) * 16;
                int ni = (t % tn) * 16;
                v16h a  = frag_a(&Qt[cur][mi + l15][0], lane);
                v16h bb = frag_b(&Vt[cur][ni + l15][0], lane);
                accs[i] = wmma_f16(a, bb, accs[i]);
            }
        }

        if (more) store_lds(cur ^ 1);
        __syncthreads();
    }

    float* Sb = S + (size_t)bh * hd * hd;
    #pragma unroll
    for (int i = 0; i < 8; ++i) {
        int t = wave + i * 8;
        if (t < tiles) {
            int mi = (t / tn) * 16;
            int ni = (t % tn) * 16;
            int n = ni + l15;
            #pragma unroll
            for (int r = 0; r < 8; ++r) {
                int m = mi + ((lane < 16) ? r : 8 + r);
                Sb[(size_t)m * hd + n] = accs[i][r] * scale;
            }
        }
    }
}

// ===========================================================================
// Attention stage 2: ATT_h = QK_h @ S[b,h]   (Ntok x hd, K = hd), f16 out.
// S staged transposed + zero-padded to K multiple of 32; A rows staged
// zero-padded. WMMA loop has no per-element guards.
// ===========================================================================
__global__ void attn_qk_s_kernel(const _Float16* __restrict__ QK,
                                 const float* __restrict__ S,
                                 _Float16* __restrict__ ATT,
                                 int B, int Nt, int D, int heads, int hd)
{
    __shared__ _Float16 St[128][136];  // [n][k], 272B row stride
    __shared__ _Float16 Aq[64][136];   // [m][k]

    const int bh = blockIdx.z;
    const int b  = bh / heads;
    const int h  = bh - b * heads;
    const int tid  = threadIdx.x;
    const int wave = tid >> 5;
    const int lane = tid & 31;
    const int l15  = lane & 15;
    const int m0 = blockIdx.x * 64;
    const int n0 = blockIdx.y * 32;
    const int wm = (wave >> 1) * 16;
    const int wn = (wave & 1) * 16;

    const int KP = (hd + 31) & ~31;
    const int NP = KP;

    const _Float16* Qb = QK + (size_t)b * Nt * D + h * hd;
    const float*    Sb = S  + (size_t)bh * hd * hd;

    // phase 1: zero-fill padded S region
    for (int i = tid; i < NP * (KP >> 3); i += 256) {
        int n   = i / (KP >> 3);
        int off = (i - n * (KP >> 3)) * 8;
        v8h z = {};
        *(v8h*)&St[n][off] = z;
    }
    __syncthreads();

    // phase 2: stage S transposed (f32->f16) + A rows (f16, zero-padded)
    for (int i = tid; i < hd * (hd >> 2); i += 256) {
        int k  = i / (hd >> 2);
        int cc = (i - k * (hd >> 2)) * 4;
        float4 f = *(const float4*)(Sb + (size_t)k * hd + cc);
        St[cc + 0][k] = (_Float16)f.x; St[cc + 1][k] = (_Float16)f.y;
        St[cc + 2][k] = (_Float16)f.z; St[cc + 3][k] = (_Float16)f.w;
    }
    for (int i = tid; i < 64 * (KP >> 3); i += 256) {
        int r  = i / (KP >> 3);
        int cc = (i - r * (KP >> 3)) * 8;
        int m  = m0 + r;
        v8h hv = {};
        if (m < Nt && cc < hd)
            hv = *(const v8h*)(Qb + (size_t)m * D + cc);
        *(v8h*)&Aq[r][cc] = hv;
    }
    __syncthreads();

    // WMMA main loop: unguarded LDS reads
    v8f acc = {};
    for (int k0 = 0; k0 < KP; k0 += 32) {
        v16h a  = frag_a(&Aq[wm + l15][k0], lane);
        v16h bb = frag_b(&St[n0 + wn + l15][k0], lane);
        acc = wmma_f16(a, bb, acc);
    }

    const int n = n0 + wn + l15;
    if (n < hd) {
        #pragma unroll
        for (int r = 0; r < 8; ++r) {
            int m = m0 + wm + ((lane < 16) ? r : 8 + r);
            if (m < Nt)
                ATT[(size_t)(b * Nt + m) * D + h * hd + n] = (_Float16)acc[r];
        }
    }
}

// ===========================================================================
// LayerNorm: one wave per row, f32 in -> f16 out (feeds GEMM A operands)
// ===========================================================================
__global__ void layernorm_kernel(const float* __restrict__ X,
                                 const float* __restrict__ g,
                                 const float* __restrict__ bt,
                                 _Float16* __restrict__ Y, int M, int D)
{
    const int row = blockIdx.x * (blockDim.x >> 5) + (threadIdx.x >> 5);
    if (row >= M) return;
    const int lane = threadIdx.x & 31;
    const float* xr = X + (size_t)row * D;
    float s = 0.0f, s2 = 0.0f;
    for (int i = lane; i < D; i += 32) { float v = xr[i]; s += v; s2 += v * v; }
    #pragma unroll
    for (int o = 16; o > 0; o >>= 1) {
        s  += __shfl_xor(s,  o, 32);
        s2 += __shfl_xor(s2, o, 32);
    }
    const float mu  = s / (float)D;
    const float var = s2 / (float)D - mu * mu;
    const float inv = rsqrtf(var + 1e-5f);
    _Float16* yr = Y + (size_t)row * D;
    for (int i = lane; i < D; i += 32)
        yr[i] = (_Float16)((xr[i] - mu) * inv * g[i] + bt[i]);
}

// ===========================================================================
// RoPE, f16 in -> f16 out (reference's double-trig variant)
// ===========================================================================
__global__ void rope_kernel(const _Float16* __restrict__ QK,
                            _Float16* __restrict__ QKR,
                            int M, int Nt, int D, int heads, int hd)
{
    const int pairs = hd >> 1;
    const int total = M * heads * pairs;
    int idx = blockIdx.x * blockDim.x + threadIdx.x;
    if (idx >= total) return;
    const int i = idx % pairs;
    const int h = (idx / pairs) % heads;
    const int m = idx / (pairs * heads);
    const float pos = (float)(m % Nt);

    const _Float16* src = QK  + (size_t)m * D + h * hd;
    _Float16*       dst = QKR + (size_t)m * D + h * hd;
    const float x1 = (float)src[2 * i];
    const float x2 = (float)src[2 * i + 1];

    const float lf = 9.210340372f;  // ln(10000)
    const float b0 = cosf(pos * __expf(-lf * (float)(4 * i)     / (float)D));
    const float b1 = cosf(pos * __expf(-lf * (float)(4 * i + 2) / (float)D));
    dst[i]         = (_Float16)(x1 * cosf(b0) - x2 * sinf(b0));
    dst[pairs + i] = (_Float16)(x1 * sinf(b1) + x2 * cosf(b1));
}

// ===========================================================================
// Conv (VALID, square kernel) with optional folded BatchNorm epilogue
// ===========================================================================
__global__ void conv_bn_kernel(const float* __restrict__ in,
                               const float* __restrict__ w,
                               const float* __restrict__ bias,
                               float* __restrict__ out,
                               int B, int Cin, int Hin, int Win,
                               int Cout, int KH, int stride,
                               const float* bn_g, const float* bn_b,
                               const float* bn_m, const float* bn_v)
{
    const int Ho = (Hin - KH) / stride + 1;
    const int Wo = (Win - KH) / stride + 1;
    int idx = blockIdx.x * blockDim.x + threadIdx.x;
    const int total = B * Cout * Ho * Wo;
    if (idx >= total) return;
    int wo = idx % Wo; int t = idx / Wo;
    int ho = t % Ho;  t /= Ho;
    int co = t % Cout;
    int b  = t / Cout;

    float acc = bias[co];
    const float* wp = w + (size_t)co * Cin * KH * KH;
    for (int ci = 0; ci < Cin; ++ci) {
        const float* ip = in + ((size_t)(b * Cin + ci) * Hin + ho * stride) * Win
                             + wo * stride;
        for (int kh = 0; kh < KH; ++kh)
            for (int kw = 0; kw < KH; ++kw)
                acc += ip[kh * Win + kw] * wp[(ci * KH + kh) * KH + kw];
    }
    if (bn_g) {
        float inv = bn_g[co] * rsqrtf(bn_v[co] + 1e-5f);
        acc = acc * inv + (bn_b[co] - bn_m[co] * inv);
    }
    out[idx] = acc;
}

// ===========================================================================
// Layout shuffles: NCHW <-> [B*N, C] token-major
// ===========================================================================
__global__ void nchw_to_tokens_kernel(const float* __restrict__ xs,
                                      float* __restrict__ xf,
                                      int B, int C, int Nt)
{
    size_t idx = (size_t)blockIdx.x * blockDim.x + threadIdx.x;
    size_t total = (size_t)B * C * Nt;
    if (idx >= total) return;
    int c = (int)(idx % C);
    size_t t = idx / C;
    int n = (int)(t % Nt);
    int b = (int)(t / Nt);
    xf[idx] = xs[((size_t)b * C + c) * Nt + n];
}

__global__ void tokens_to_nchw_kernel(const float* __restrict__ xf,
                                      float* __restrict__ xs,
                                      int B, int C, int Nt)
{
    size_t idx = (size_t)blockIdx.x * blockDim.x + threadIdx.x;
    size_t total = (size_t)B * C * Nt;
    if (idx >= total) return;
    int n = (int)(idx % Nt);
    size_t t = idx / Nt;
    int c = (int)(t % C);
    int b = (int)(t / C);
    xs[idx] = xf[((size_t)b * Nt + n) * C + c];
}

// ===========================================================================
// Host orchestration
// ===========================================================================
extern "C" void kernel_launch(void* const* d_in, const int* in_sizes, int n_in,
                              void* d_out, int out_size, void* d_ws, size_t ws_size,
                              hipStream_t stream)
{
    (void)in_sizes; (void)n_in; (void)out_size; (void)ws_size;
    const int B = 4, HEADS = 4;
    const int DIMS[4]   = {64, 128, 256, 512};
    const int DEPTHS[4] = {2, 2, 4, 2};

    const size_t SZ = 802816;            // max token elements = 4*3136*64
    float* XS   = (float*)d_ws;          // spatial NCHW (f32)
    float* XF   = XS   + SZ;             // residual stream (f32)
    float* GATE = XF   + SZ;             // gate (f32)
    float* Sbuf = GATE + SZ;             // B*H*hd*hd (f32, max 262144)
    _Float16* Hb   = (_Float16*)(Sbuf + 262144);
    _Float16* XNh  = Hb;                 // layernorm out (f16)
    _Float16* XLh  = XNh  + SZ;          // relu(enc)     (f16)
    _Float16* QKh  = XLh  + SZ;
    _Float16* QKRh = QKh  + SZ;
    _Float16* Vh   = QKRh + SZ;
    _Float16* ATTh = Vh   + SZ;
    _Float16* WHb  = ATTh + SZ;          // f16 weight arena (~8.3 MB)

    auto in = [&](int i) { return (const float*)d_in[i]; };

    const int BLK_BASE  = 3;
    const int DOWN_BASE = 3 + 12 * (2 + 2 + 4 + 2);
    const int WIDX[5]   = {2, 4, 6, 8, 10};   // enc_w, qk_w, v_w, out_w, gate_w

    // ---- convert all block weights to f16 once ----
    _Float16* whp[10][5];
    {
        _Float16* wh = WHb;
        int blk2 = 0;
        for (int s2 = 0; s2 < 4; ++s2) {
            int Dd = DIMS[s2];
            for (int bl = 0; bl < DEPTHS[s2]; ++bl, ++blk2) {
                const int p = BLK_BASE + blk2 * 12;
                for (int wi = 0; wi < 5; ++wi) {
                    int n4 = Dd * Dd / 4;
                    cvt_f16_kernel<<<(n4 + 255) / 256, 256, 0, stream>>>(
                        in(p + WIDX[wi]), wh, n4);
                    whp[blk2][wi] = wh;
                    wh += (size_t)Dd * Dd;
                }
            }
        }
    }

    // ---- patch embedding: (4,3,224,224) -> XS (4,64,56,56) ----
    {
        int total = B * 64 * 56 * 56;
        conv_bn_kernel<<<(total + 255) / 256, 256, 0, stream>>>(
            in(0), in(1), in(2), XS, B, 3, 224, 224, 64, 4, 4,
            nullptr, nullptr, nullptr, nullptr);
    }

    float* out = (float*)d_out;
    size_t feat_off = 0;
    int H = 56;
    int blk = 0;

    for (int s = 0; s < 4; ++s) {
        const int D  = DIMS[s];
        const int Nt = H * H;
        const int M  = B * Nt;
        const int hd = D / HEADS;
        const float scale = 1.0f / sqrtf((float)hd);

        {
            size_t total = (size_t)B * D * Nt;
            nchw_to_tokens_kernel<<<(int)((total + 255) / 256), 256, 0, stream>>>(
                XS, XF, B, D, Nt);
        }

        for (int bl = 0; bl < DEPTHS[s]; ++bl, ++blk) {
            const int p = BLK_BASE + blk * 12;
            const float* ln_g   = in(p + 0);
            const float* ln_b   = in(p + 1);
            const float* enc_b  = in(p + 3);
            const float* qk_b   = in(p + 5);
            const float* v_b    = in(p + 7);
            const float* out_b  = in(p + 9);
            const float* gate_b = in(p + 11);

            layernorm_kernel<<<(M + 7) / 8, 256, 0, stream>>>(XF, ln_g, ln_b, XNh, M, D);

            dim3 gg((M + 127) / 128, D / 64);
            // xl = relu(xn @ enc_w^T + b)  (f16 out)
            wmma_gemm_kernel<<<gg, 256, 0, stream>>>(XNh, whp[blk][0], enc_b,
                nullptr, XLh, M, D, D, 1, nullptr, nullptr);
            // qk / v (f16 out)
            wmma_gemm_kernel<<<gg, 256, 0, stream>>>(XLh, whp[blk][1], qk_b,
                nullptr, QKh, M, D, D, 0, nullptr, nullptr);
            wmma_gemm_kernel<<<gg, 256, 0, stream>>>(XLh, whp[blk][2], v_b,
                nullptr, Vh, M, D, D, 0, nullptr, nullptr);

            {
                int total = M * (D / 2);
                rope_kernel<<<(total + 255) / 256, 256, 0, stream>>>(
                    QKh, QKRh, M, Nt, D, HEADS, hd);
            }
            attn_qkT_v_kernel<<<B * HEADS, 256, 0, stream>>>(
                QKRh, Vh, Sbuf, B, Nt, D, HEADS, hd, scale);
            {
                dim3 g2((Nt + 63) / 64, (hd + 31) / 32, B * HEADS);
                attn_qk_s_kernel<<<g2, 256, 0, stream>>>(
                    QKRh, Sbuf, ATTh, B, Nt, D, HEADS, hd);
            }
            // gate = sigmoid(xn @ gate_w^T + b)  (f32 out)
            wmma_gemm_kernel<<<gg, 256, 0, stream>>>(XNh, whp[blk][4], gate_b,
                GATE, nullptr, M, D, D, 2, nullptr, nullptr);
            // xf = xf + gate * (attn @ out_w^T + b)
            wmma_gemm_kernel<<<gg, 256, 0, stream>>>(ATTh, whp[blk][3], out_b,
                XF, nullptr, M, D, D, 0, GATE, XF);
        }

        {
            size_t total = (size_t)B * D * Nt;
            tokens_to_nchw_kernel<<<(int)((total + 255) / 256), 256, 0, stream>>>(
                XF, out + feat_off, B, D, Nt);
        }

        if (s < 3) {
            const int q = DOWN_BASE + s * 6;
            const int C2 = DIMS[s + 1];
            int total = B * C2 * (H / 2) * (H / 2);
            conv_bn_kernel<<<(total + 255) / 256, 256, 0, stream>>>(
                out + feat_off, in(q + 0), in(q + 1), XS,
                B, D, H, H, C2, 2, 2,
                in(q + 2), in(q + 3), in(q + 4), in(q + 5));
        }

        feat_off += (size_t)B * D * Nt;
        H >>= 1;
    }
}
